// QCCPResidualBlock_87514253623728
// MI455X (gfx1250) — matrix operand order repaired and checked by reading, not compiled, not run on hardware
//
#include <hip/hip_runtime.h>

// ---------------------------------------------------------------------------
// CDNA5 (gfx1250) fused ResNet basic block:
//   y1 = conv3x3(x,w1)+b1            (WMMA bf16 implicit GEMM)
//   yp = conv1x1(x,wp)+bp            (fused into conv1's center tap)
//   bn1 stats -> scale/shift; conv2 input = relu(bn1(y1)) fused into staging
//   y2 = conv3x3(.,w2)+b2
//   out = relu(bn2(y2) + bnp(yp))
// Weights live in LDS, filled via GLOBAL_LOAD_ASYNC_TO_LDS (ASYNCcnt path).
// ---------------------------------------------------------------------------

typedef __attribute__((ext_vector_type(16))) __bf16 v16bf;
typedef __attribute__((ext_vector_type(8)))  __bf16 v8bf;
typedef __attribute__((ext_vector_type(8)))  float  v8f;
typedef __attribute__((ext_vector_type(4)))  int    v4i;

#define H_   512
#define W_   512
#define C_   64
#define N_   2
#define HW_  (H_*W_)
#define K3   576            // 9 taps * 64 cin, k = tap*64 + cin
#define YSZ  33554432       // N_*C_*H_*W_ elements
#define CNT_ 524288.0f      // N_*H_*W_ (per-channel BN count)
#define EPS_ 1e-5f

// LDS x tile: 6 halo rows x 34 halo cols, cin-contiguous, pixel stride padded
// to 72 bf16 (=36 dwords; gcd(36,64)=4 -> conflict-free b128 bank pattern)
#define TROW 6
#define TCOL 34
#define PSTR 72
// LDS weight copies: rows padded for the same conflict-free pattern
#define ASTR 584            // 3x3 weights: 576 -> 584 (292 dwords, mod 64 = 36)
#define APSTR 72            // 1x1 weights: 64 -> 72
#define XS_BYTES   (TROW*TCOL*PSTR*2)          // 29376
#define AS_BYTES   (64*ASTR*2)                 // 74752
#define APS_BYTES  (64*APSTR*2)                // 9216
#define SMEM_PROJ  (XS_BYTES + AS_BYTES + APS_BYTES)   // 113344
#define SMEM_BASE  (XS_BYTES + AS_BYTES)               // 104128

#if defined(__gfx1250__) && \
    __has_builtin(__builtin_amdgcn_global_load_async_to_lds_b128) && \
    __has_builtin(__builtin_amdgcn_s_wait_asynccnt)
#define USE_ASYNC_LDS 1
#else
#define USE_ASYNC_LDS 0
#endif

static __device__ __forceinline__ v8f wmma_bf16(v16bf a, v16bf b, v8f c) {
  return __builtin_amdgcn_wmma_f32_16x16x32_bf16(false, a, false, b,
                                                 (short)0, c, false, false);
}

static __device__ __forceinline__ v16bf cat8(v8bf lo, v8bf hi) {
  return __builtin_shufflevector(lo, hi, 0,1,2,3,4,5,6,7,
                                         8,9,10,11,12,13,14,15);
}

// copy one 16-byte chunk global -> LDS
static __device__ __forceinline__ void copy16_g2l(const __bf16* g, __bf16* l) {
#if USE_ASYNC_LDS
  __builtin_amdgcn_global_load_async_to_lds_b128(
      (__attribute__((address_space(1))) v4i*)g,
      (__attribute__((address_space(3))) v4i*)l, 0, 0);
#else
  *(v8bf*)l = *(const v8bf*)g;
#endif
}

static __device__ __forceinline__ void copy_wait() {
#if USE_ASYNC_LDS
  __builtin_amdgcn_s_wait_asynccnt(0);
#endif
}

// ---- pack OIHW fp32 weights -> bf16 A[m][k], k = tap*64 + cin -------------
__global__ __launch_bounds__(256) void pack_w3(const float* __restrict__ w,
                                               __bf16* __restrict__ A) {
  int idx = blockIdx.x * 256 + threadIdx.x;          // 64*576
  if (idx >= 64 * K3) return;
  int m = idx / K3, k = idx % K3;
  int r = k >> 6, c = k & 63;
  A[idx] = (__bf16)w[(m * 64 + c) * 9 + r];
}

__global__ __launch_bounds__(256) void pack_w1(const float* __restrict__ w,
                                               __bf16* __restrict__ A) {
  int idx = blockIdx.x * 256 + threadIdx.x;          // 64*64
  if (idx >= 64 * 64) return;
  A[idx] = (__bf16)w[idx];
}

__global__ __launch_bounds__(256) void zero_kernel(float* __restrict__ p, int n) {
  int i = blockIdx.x * 256 + threadIdx.x;
  if (i < n) p[i] = 0.f;
}

// ---- fused implicit-GEMM conv3x3 (+optional 1x1 projection on center tap) -
// block: 256 threads (8 waves), output tile = 64 cout x (4 rows x 32 cols)
template <bool BNIN, bool PROJ>
__global__ __launch_bounds__(256, 2) void conv3x3_wmma(
    const float*  __restrict__ xin,
    const __bf16* __restrict__ A,      // [64][576] bf16 packed
    const float*  __restrict__ bias,
    const __bf16* __restrict__ Ap,     // [64][64] bf16 packed (PROJ)
    const float*  __restrict__ biasp,
    const float*  __restrict__ sc_in,  // per-cin BN scale (BNIN)
    const float*  __restrict__ sh_in,  // per-cin BN shift (BNIN)
    float* __restrict__ yout,
    float* __restrict__ ypout) {
  extern __shared__ char smem[];
  __bf16* xs   = (__bf16*)smem;                       // x tile
  __bf16* Asm  = (__bf16*)(smem + XS_BYTES);          // 3x3 weights
  __bf16* Apsm = (__bf16*)(smem + XS_BYTES + AS_BYTES);  // 1x1 weights

  const int tid = threadIdx.x;
  const int t   = blockIdx.x;          // 4096 tiles
  const int n   = t >> 11;             // / 2048
  const int rem = t & 2047;
  const int ty0 = (rem >> 4) << 2;     // 4-row tiles
  const int tx0 = (rem & 15) << 5;     // 32-col tiles

  // ---- async-fill weight LDS: A is 64 rows x 72 chunks of 16B ------------
  for (int i = tid; i < 64 * 72; i += 256) {
    int row = i / 72, ck = i % 72;
    copy16_g2l(A + row * K3 + ck * 8, Asm + row * ASTR + ck * 8);
  }
  if constexpr (PROJ) {
    for (int i = tid; i < 64 * 8; i += 256) {   // 64 rows x 8 chunks
      int row = i >> 3, ck = i & 7;
      copy16_g2l(Ap + row * 64 + ck * 8, Apsm + row * APSTR + ck * 8);
    }
  }

  // ---- stage halo x tile to LDS, transposed to [row][col][cin], bf16 -----
  for (int c = 0; c < C_; ++c) {
    const float* src = xin + (size_t)(n * C_ + c) * HW_;
    float sc = 1.f, sh = 0.f;
    if constexpr (BNIN) { sc = sc_in[c]; sh = sh_in[c]; }
    int e = tid;
    if (e < TROW * TCOL) {
      int r = e / TCOL, col = e % TCOL;
      int gy = ty0 + r - 1, gx = tx0 + col - 1;
      float v = 0.f;
      bool inb = (gy >= 0) & (gy < H_) & (gx >= 0) & (gx < W_);
      if (inb) {
        const float* ap = src + gy * W_ + gx;
        v = *ap;
        if (c + 1 < C_) __builtin_prefetch(ap + HW_, 0, 1);  // global_prefetch_b8
      }
      if constexpr (BNIN) { v = v * sc + sh; v = v > 0.f ? v : 0.f; }
      xs[e * PSTR + c] = (__bf16)v;
    }
  }
  copy_wait();          // this wave's async weight copies complete
  __syncthreads();      // everyone's copies + ds stores visible

  const int lane = tid & 31;
  const int l16  = lane & 15;
  const int lhi  = lane >> 4;          // which 16-lane half
  const int wv   = tid >> 5;           // wave 0..7
  const int mt0  = (wv & 1) << 1;      // cout tiles {0,1} or {2,3}
  const int nt0  = (wv >> 1) << 1;     // pixel tiles (8 total per block)

  v8f acc[2][2];
  v8f accp[2][2];
  for (int mi = 0; mi < 2; ++mi) {
    v8f bv;
    for (int j = 0; j < 8; ++j) bv[j] = bias[(mt0 + mi) * 16 + j + 8 * lhi];
    acc[mi][0] = bv; acc[mi][1] = bv;
    if constexpr (PROJ) {
      v8f pv;
      for (int j = 0; j < 8; ++j) pv[j] = biasp[(mt0 + mi) * 16 + j + 8 * lhi];
      accp[mi][0] = pv; accp[mi][1] = pv;
    }
  }

  // local pixel coordinates for this lane's two N-tiles
  const int q0 = nt0 * 16 + l16,        q1 = (nt0 + 1) * 16 + l16;
  const int ly0 = q0 >> 5, lx0 = q0 & 31;
  const int ly1 = q1 >> 5, lx1 = q1 & 31;

#pragma unroll
  for (int ks = 0; ks < 18; ++ks) {
    const int k0 = ks << 5;            // K step of 32
    const int r  = k0 >> 6;            // filter tap 0..8 (fixed per step)
    const int ry = r / 3, rx = r % 3;
    const int c0 = (k0 & 32) + lhi * 8;   // cin base for B fragment

    // A fragments from LDS: lane half holds K chunks {lhi*8, 16+lhi*8}
    const __bf16* ar0 = Asm + ((mt0    ) * 16 + l16) * ASTR + k0;
    const __bf16* ar1 = Asm + ((mt0 + 1) * 16 + l16) * ASTR + k0;
    v16bf a0 = cat8(*(const v8bf*)(ar0 + lhi * 8),
                    *(const v8bf*)(ar0 + 16 + lhi * 8));
    v16bf a1 = cat8(*(const v8bf*)(ar1 + lhi * 8),
                    *(const v8bf*)(ar1 + 16 + lhi * 8));

    // B fragments: cin chunks {lhi*8, 16+lhi*8} from LDS (two ds_load_b128 each)
    const __bf16* b0p = &xs[((ly0 + ry) * TCOL + lx0 + rx) * PSTR + c0];
    const __bf16* b1p = &xs[((ly1 + ry) * TCOL + lx1 + rx) * PSTR + c0];
    v16bf b0 = cat8(*(const v8bf*)(b0p), *(const v8bf*)(b0p + 16));
    v16bf b1 = cat8(*(const v8bf*)(b1p), *(const v8bf*)(b1p + 16));

    acc[0][0] = wmma_bf16(a0, b0, acc[0][0]);
    acc[0][1] = wmma_bf16(a0, b1, acc[0][1]);
    acc[1][0] = wmma_bf16(a1, b0, acc[1][0]);
    acc[1][1] = wmma_bf16(a1, b1, acc[1][1]);

    if constexpr (PROJ) {
      if (r == 4) {  // center tap: same B fragments drive the 1x1 projection
        const __bf16* ap0 = Apsm + ((mt0    ) * 16 + l16) * APSTR + (k0 & 32);
        const __bf16* ap1 = Apsm + ((mt0 + 1) * 16 + l16) * APSTR + (k0 & 32);
        v16bf pa0 = cat8(*(const v8bf*)(ap0 + lhi * 8),
                         *(const v8bf*)(ap0 + 16 + lhi * 8));
        v16bf pa1 = cat8(*(const v8bf*)(ap1 + lhi * 8),
                         *(const v8bf*)(ap1 + 16 + lhi * 8));
        accp[0][0] = wmma_bf16(pa0, b0, accp[0][0]);
        accp[0][1] = wmma_bf16(pa0, b1, accp[0][1]);
        accp[1][0] = wmma_bf16(pa1, b0, accp[1][0]);
        accp[1][1] = wmma_bf16(pa1, b1, accp[1][1]);
      }
    }
  }

  // ---- store NCHW fp32 (D layout: vgpr j, lane -> M=j+8*lhi, N=l16) ------
  // one 32-bit index + base pointer per tile; per-j channel stride j*HW_
  // (j*1MB <= 7MB) folds into the signed 24-bit global immediate offset.
#pragma unroll
  for (int mi = 0; mi < 2; ++mi) {
#pragma unroll
    for (int ni = 0; ni < 2; ++ni) {
      const int ly = ni ? ly1 : ly0, lx = ni ? lx1 : lx0;
      const int idx = ((n * C_ + (mt0 + mi) * 16 + 8 * lhi) * H_ + ty0 + ly) * W_
                      + tx0 + lx;
      float* dst = yout + idx;
#pragma unroll
      for (int j = 0; j < 8; ++j) dst[j * HW_] = acc[mi][ni][j];
      if constexpr (PROJ) {
        float* dstp = ypout + idx;
#pragma unroll
        for (int j = 0; j < 8; ++j) dstp[j * HW_] = accp[mi][ni][j];
      }
    }
  }
}

// ---- per-channel sum / sumsq over (N,H,W) ---------------------------------
__global__ __launch_bounds__(256) void stats_kernel(const float* __restrict__ y,
                                                    float* __restrict__ sums,
                                                    float* __restrict__ sumsq) {
  const int c = blockIdx.y;
  float s = 0.f, s2 = 0.f;
  for (int n = 0; n < N_; ++n) {
    const float* p = y + (size_t)(n * C_ + c) * HW_;
    for (int i = blockIdx.x * 256 + threadIdx.x; i < HW_; i += gridDim.x * 256) {
      float v = p[i];
      s += v; s2 += v * v;
    }
  }
  __shared__ float rs[256], rq[256];
  rs[threadIdx.x] = s; rq[threadIdx.x] = s2;
  __syncthreads();
  for (int off = 128; off > 0; off >>= 1) {
    if (threadIdx.x < off) {
      rs[threadIdx.x] += rs[threadIdx.x + off];
      rq[threadIdx.x] += rq[threadIdx.x + off];
    }
    __syncthreads();
  }
  if (threadIdx.x == 0) {
    atomicAdd(&sums[c], rs[0]);
    atomicAdd(&sumsq[c], rq[0]);
  }
}

__global__ void finalize_scale(const float* __restrict__ sums,
                               const float* __restrict__ sumsq,
                               const float* __restrict__ g,
                               const float* __restrict__ be,
                               float* __restrict__ scale,
                               float* __restrict__ shift) {
  int c = threadIdx.x;
  if (c >= C_) return;
  float mean = sums[c] * (1.0f / CNT_);
  float var  = sumsq[c] * (1.0f / CNT_) - mean * mean;
  float sc   = g[c] * rsqrtf(var + EPS_);
  scale[c] = sc;
  shift[c] = be[c] - mean * sc;
}

// ---- out = relu(bn2(y2) + bnp(yp)), float4 vectorized ---------------------
__global__ __launch_bounds__(256) void final_fuse(
    const float* __restrict__ y2, const float* __restrict__ yp,
    const float* __restrict__ sc2, const float* __restrict__ sh2,
    const float* __restrict__ scp, const float* __restrict__ shp,
    float* __restrict__ out) {
  const int i4 = blockIdx.x * 256 + threadIdx.x;   // YSZ/4 = 8388608
  if (i4 >= YSZ / 4) return;
  const int c = (i4 >> 16) & 63;                   // HW_/4 = 65536
  float4 a = ((const float4*)y2)[i4];
  float4 b = ((const float4*)yp)[i4];
  float s2 = sc2[c], h2 = sh2[c], sp = scp[c], hp = shp[c];
  float4 o;
  o.x = fmaxf(a.x * s2 + h2 + b.x * sp + hp, 0.f);
  o.y = fmaxf(a.y * s2 + h2 + b.y * sp + hp, 0.f);
  o.z = fmaxf(a.z * s2 + h2 + b.z * sp + hp, 0.f);
  o.w = fmaxf(a.w * s2 + h2 + b.w * sp + hp, 0.f);
  ((float4*)out)[i4] = o;
}

// ---------------------------------------------------------------------------
extern "C" void kernel_launch(void* const* d_in, const int* in_sizes, int n_in,
                              void* d_out, int out_size, void* d_ws, size_t ws_size,
                              hipStream_t stream) {
  const float* x   = (const float*)d_in[0];
  const float* w1  = (const float*)d_in[1];
  const float* b1  = (const float*)d_in[2];
  const float* g1  = (const float*)d_in[3];
  const float* be1 = (const float*)d_in[4];
  const float* w2  = (const float*)d_in[5];
  const float* b2  = (const float*)d_in[6];
  const float* g2  = (const float*)d_in[7];
  const float* be2 = (const float*)d_in[8];
  const float* wp  = (const float*)d_in[9];
  const float* bp  = (const float*)d_in[10];
  const float* gp  = (const float*)d_in[11];
  const float* bep = (const float*)d_in[12];
  float* out = (float*)d_out;

  char* ws = (char*)d_ws;
  float*  y1  = (float*)ws;   ws += (size_t)YSZ * 4;
  float*  y2  = (float*)ws;   ws += (size_t)YSZ * 4;
  float*  yp  = (float*)ws;   ws += (size_t)YSZ * 4;
  __bf16* A1  = (__bf16*)ws;  ws += (size_t)64 * K3 * 2;
  __bf16* A2  = (__bf16*)ws;  ws += (size_t)64 * K3 * 2;
  __bf16* Apj = (__bf16*)ws;  ws += (size_t)64 * 64 * 2;
  float* stat = (float*)ws;   // 12 x 64 floats
  float *s1 = stat,       *q1 = stat + 64,  *s2 = stat + 128, *q2 = stat + 192;
  float *sp = stat + 256, *qp = stat + 320;
  float *sc1 = stat + 384, *sh1 = stat + 448;
  float *sc2 = stat + 512, *sh2 = stat + 576;
  float *scp = stat + 640, *shp = stat + 704;

  zero_kernel<<<2, 256, 0, stream>>>(stat, 384);
  pack_w3<<<144, 256, 0, stream>>>(w1, A1);
  pack_w3<<<144, 256, 0, stream>>>(w2, A2);
  pack_w1<<<16, 256, 0, stream>>>(wp, Apj);

  // conv1 + fused 1x1 projection
  conv3x3_wmma<false, true><<<4096, 256, SMEM_PROJ, stream>>>(
      x, A1, b1, Apj, bp, nullptr, nullptr, y1, yp);

  stats_kernel<<<dim3(32, 64), 256, 0, stream>>>(y1, s1, q1);
  finalize_scale<<<1, 64, 0, stream>>>(s1, q1, g1, be1, sc1, sh1);

  // conv2 with bn1+relu fused into input staging
  conv3x3_wmma<true, false><<<4096, 256, SMEM_BASE, stream>>>(
      y1, A2, b2, nullptr, nullptr, sc1, sh1, y2, nullptr);

  stats_kernel<<<dim3(32, 64), 256, 0, stream>>>(y2, s2, q2);
  stats_kernel<<<dim3(32, 64), 256, 0, stream>>>(yp, sp, qp);
  finalize_scale<<<1, 64, 0, stream>>>(s2, q2, g2, be2, sc2, sh2);
  finalize_scale<<<1, 64, 0, stream>>>(sp, qp, gp, bep, scp, shp);

  final_fuse<<<YSZ / 4 / 256, 256, 0, stream>>>(y2, yp, sc2, sh2, scp, shp, out);

  (void)in_sizes; (void)n_in; (void)out_size; (void)ws_size;
}
